// Network_31636729103176
// MI455X (gfx1250) — compile-verified, matrix operand order checked by reading, not run on hardware
//
#include <hip/hip_runtime.h>
#include <math.h>

// ---------------------------------------------------------------------------
// Plastic RNN step for MI455X (gfx1250, wave32).
// Memory-bound (~320MB @ 23.3TB/s ~ 14us). f32 GEMV kept exact; hebb rank-1
// update done per-16x16 tile with v_wmma_f32_16x16x32_f16 (f16 error only on
// the eta*outer term). NT loads on w/alpha keep hebb resident in the 192MB L2
// between the GEMV and the hebb-update kernel.
// ---------------------------------------------------------------------------

typedef float v4f  __attribute__((ext_vector_type(4)));
typedef float v8f  __attribute__((ext_vector_type(8)));
typedef _Float16 v16h __attribute__((ext_vector_type(16)));

#define HDIM   4096
#define NIN    17
#define NA     4
#define CHUNK  32                 // rows of the H x H matrices per split-K block
#define NCHUNK (HDIM / CHUNK)     // 128 partial-sum slabs

// ---------------------------------------------------------------------------
// Kernel A: split-K GEMV over the combined matrix (w + alpha*hebb).
// grid = (HDIM/(256*4), NCHUNK) = (4, 128) blocks of 256 threads.
// Each thread owns 4 consecutive columns (b128 loads), loops CHUNK rows.
// w/alpha streamed with NT hint; hebb with default RT hint (stays in L2).
// ---------------------------------------------------------------------------
__global__ void k_gemv(const float* __restrict__ hidden,
                       const float* __restrict__ hebb,
                       const float* __restrict__ w,
                       const float* __restrict__ alpha,
                       float* __restrict__ zpart) {
  __shared__ float sh[CHUNK];
  const int col = (blockIdx.x * blockDim.x + threadIdx.x) * 4;
  const int i0  = blockIdx.y * CHUNK;
  if (threadIdx.x < CHUNK) sh[threadIdx.x] = hidden[i0 + threadIdx.x];
  __syncthreads();

  v4f acc = {0.f, 0.f, 0.f, 0.f};
  for (int r = 0; r < CHUNK; ++r) {
    const size_t off = (size_t)(i0 + r) * HDIM + col;
    v4f wv = __builtin_nontemporal_load((const v4f*)(w + off));      // NT: don't pollute L2
    v4f av = __builtin_nontemporal_load((const v4f*)(alpha + off));  // NT: don't pollute L2
    v4f hb = *(const v4f*)(hebb + off);                              // RT: keep in L2 for k_hebb
    const float hv = sh[r];
    acc += hv * (wv + av * hb);
  }
  *(v4f*)(zpart + (size_t)blockIdx.y * HDIM + col) = acc;
}

// ---------------------------------------------------------------------------
// Kernel B: deterministic reduce of the 128 partials + i2h + bias, tanh -> h.
// ---------------------------------------------------------------------------
__global__ void k_act(const float* __restrict__ zpart,
                      const float* __restrict__ x,
                      const float* __restrict__ i2h_w,
                      const float* __restrict__ i2h_b,
                      float* __restrict__ out_h) {
  const int j = blockIdx.x * blockDim.x + threadIdx.x;
  float s = i2h_b[j];
#pragma unroll
  for (int k = 0; k < NIN; ++k) s += x[k] * i2h_w[j * NIN + k];
  float z = 0.f;
  for (int c = 0; c < NCHUNK; ++c) z += zpart[(size_t)c * HDIM + j];
  out_h[j] = tanhf(s + z);
}

// ---------------------------------------------------------------------------
// Kernel C: hebb_new = (1-eta)*hebb + eta*outer(hidden, h) via WMMA.
// One 16x16 tile per wave: D = A x B + C with
//   A[m][0] = eta*hidden[i0+m] (f16, rest 0)   -> lanes 0-15, element 0
//   B[0][n] = h[j0+n]          (f16, rest 0)   -> lanes 0-15, element 0
//   C[v]    = (1-eta)*hebb[i0 + v + 8*(lane/16)][j0 + lane%16]  (f32)
// 65536 tiles, 8 waves/block -> 8192 blocks. hebb read hits L2 (kernel A).
// ---------------------------------------------------------------------------
__global__ void k_hebb(const float* __restrict__ hebb,
                       const float* __restrict__ hidden,
                       const float* __restrict__ h,
                       const float* __restrict__ eta_p,
                       float* __restrict__ out_hebb) {
  const int lane = threadIdx.x & 31;
  const int wid  = (blockIdx.x * blockDim.x + threadIdx.x) >> 5;  // global wave id
  const int ti   = wid >> 8;        // HDIM/16 = 256 tiles per dimension
  const int tj   = wid & 255;
  const int i0   = ti * 16;
  const int j0   = tj * 16;

  const float eta    = *eta_p;
  const float one_me = 1.0f - eta;
  const int   l15    = lane & 15;

  // branch-free lane fills (EXEC stays all-ones for the WMMA)
  const float hv = hidden[i0 + l15];
  const float bv = h[j0 + l15];
  v16h A = {};
  v16h B = {};
  A[0] = (lane < 16) ? (_Float16)(eta * hv) : (_Float16)0.f;
  B[0] = (lane < 16) ? (_Float16)bv         : (_Float16)0.f;

  const size_t base = (size_t)(i0 + ((lane >> 4) << 3)) * HDIM + j0 + l15;
  v8f C;
#pragma unroll
  for (int v = 0; v < 8; ++v) C[v] = one_me * hebb[base + (size_t)v * HDIM];

  // (neg_a, A, neg_b, B, c_mod, C, reuse_a, reuse_b)
  v8f D = __builtin_amdgcn_wmma_f32_16x16x32_f16(false, A, false, B,
                                                 (short)0, C, false, false);
#pragma unroll
  for (int v = 0; v < 8; ++v) out_hebb[base + (size_t)v * HDIM] = D[v];
}

// ---------------------------------------------------------------------------
// Kernel D: heads — logits = h@h2o_w^T + b -> softmax; value = h@h2v_w^T + b.
// Single block, LDS tree reduction; trivial cost.
// ---------------------------------------------------------------------------
__global__ void k_head(const float* __restrict__ h,
                       const float* __restrict__ h2o_w,
                       const float* __restrict__ h2o_b,
                       const float* __restrict__ h2v_w,
                       const float* __restrict__ h2v_b,
                       float* __restrict__ out_act,
                       float* __restrict__ out_val) {
  __shared__ float red[5][256];
  const int t = threadIdx.x;
  float a0 = 0.f, a1 = 0.f, a2 = 0.f, a3 = 0.f, vv = 0.f;
  for (int j = t; j < HDIM; j += 256) {
    const float hv = h[j];
    a0 += hv * h2o_w[0 * HDIM + j];
    a1 += hv * h2o_w[1 * HDIM + j];
    a2 += hv * h2o_w[2 * HDIM + j];
    a3 += hv * h2o_w[3 * HDIM + j];
    vv += hv * h2v_w[j];
  }
  red[0][t] = a0; red[1][t] = a1; red[2][t] = a2; red[3][t] = a3; red[4][t] = vv;
  __syncthreads();
  for (int s = 128; s > 0; s >>= 1) {
    if (t < s) {
#pragma unroll
      for (int q = 0; q < 5; ++q) red[q][t] += red[q][t + s];
    }
    __syncthreads();
  }
  if (t == 0) {
    const float l0 = red[0][0] + h2o_b[0];
    const float l1 = red[1][0] + h2o_b[1];
    const float l2 = red[2][0] + h2o_b[2];
    const float l3 = red[3][0] + h2o_b[3];
    const float m  = fmaxf(fmaxf(l0, l1), fmaxf(l2, l3));
    const float e0 = expf(l0 - m), e1 = expf(l1 - m);
    const float e2 = expf(l2 - m), e3 = expf(l3 - m);
    const float inv = 1.0f / (e0 + e1 + e2 + e3);
    out_act[0] = e0 * inv; out_act[1] = e1 * inv;
    out_act[2] = e2 * inv; out_act[3] = e3 * inv;
    out_val[0] = red[4][0] + h2v_b[0];
  }
}

// ---------------------------------------------------------------------------
extern "C" void kernel_launch(void* const* d_in, const int* in_sizes, int n_in,
                              void* d_out, int out_size, void* d_ws, size_t ws_size,
                              hipStream_t stream) {
  const float* x      = (const float*)d_in[0];
  const float* hidden = (const float*)d_in[1];
  const float* hebb   = (const float*)d_in[2];
  const float* i2h_w  = (const float*)d_in[3];
  const float* i2h_b  = (const float*)d_in[4];
  const float* w      = (const float*)d_in[5];
  const float* alpha  = (const float*)d_in[6];
  const float* eta    = (const float*)d_in[7];
  const float* h2o_w  = (const float*)d_in[8];
  const float* h2o_b  = (const float*)d_in[9];
  const float* h2v_w  = (const float*)d_in[10];
  const float* h2v_b  = (const float*)d_in[11];

  // outputs concatenated flat in return order: activout, valueout, h, hebb_new
  float* out      = (float*)d_out;
  float* out_act  = out;
  float* out_val  = out + NA;
  float* out_h    = out + NA + 1;
  float* out_hebb = out + NA + 1 + HDIM;

  float* zpart = (float*)d_ws;  // NCHUNK * HDIM floats = 2 MB scratch

  dim3 gA(HDIM / (256 * 4), NCHUNK);                         // (4, 128)
  k_gemv<<<gA, 256, 0, stream>>>(hidden, hebb, w, alpha, zpart);
  k_act <<<HDIM / 256, 256, 0, stream>>>(zpart, x, i2h_w, i2h_b, out_h);
  k_hebb<<<(HDIM / 16) * (HDIM / 16) / 8, 256, 0, stream>>>(hebb, hidden, out_h,
                                                            eta, out_hebb);
  k_head<<<1, 256, 0, stream>>>(out_h, h2o_w, h2o_b, h2v_w, h2v_b,
                                out_act, out_val);
}
